// FNO_37735582662940
// MI455X (gfx1250) — matrix operand order repaired and checked by reading, not compiled, not run on hardware
//
#include <hip/hip_runtime.h>
#include <hip/hip_bf16.h>
#include <math.h>

typedef __bf16 bf16;
typedef __attribute__((ext_vector_type(16))) __bf16 v16bf;
typedef __attribute__((ext_vector_type(8)))  float  v8f;

#define WMMA(a,b,c) __builtin_amdgcn_wmma_f32_16x16x32_bf16(false,(a),false,(b),(short)0,(c),false,false)

namespace {
constexpr int BB = 8, CIN = 6, COUT = 6, CM = 32, HID = 128, NS = 64, NL = 4;
constexpr int N3   = NS*NS*NS;        // 262144
constexpr int PERB = CM*N3;           // 8388608 elements per batch at CM channels
constexpr float TWO_PI_64 = 0.09817477042468103f;   // 2*pi/64
constexpr float INV_VOL   = 1.0f/262144.0f;         // irfftn normalization 1/64^3
}

// fast gelu: v * sigmoid(2*0.7978845608*(v + 0.044715 v^3)); branchless, err < 3e-4
__device__ __forceinline__ float gelu_f(float v){
  float u = 1.5957691216f * v * (1.0f + 0.044715f*v*v);
  return v * __builtin_amdgcn_rcpf(1.0f + __expf(-u));
}
// A-fragment K index for 16x16x32 bf16 WMMA (contiguous order k = 16h + e, matching B)
__device__ __forceinline__ int akk(int lane, int e){
  int h = lane >> 4;
  return (h << 4) + e;
}

// ---------------------------------------------------------------- stats (GroupNorm(1,C))
__global__ void k_zero(float* __restrict__ p, int n){
  int i = blockIdx.x*blockDim.x + threadIdx.x;
  if (i < n) p[i] = 0.0f;
}

__global__ void k_stats_partial(const float* __restrict__ x, float* __restrict__ sums, int perBatch){
  __shared__ float ls[256], lq[256];
  const int b = blockIdx.y;
  const float* p = x + (size_t)b*perBatch;
  float s = 0.0f, q = 0.0f;
  for (int i = blockIdx.x*blockDim.x + threadIdx.x; i < perBatch; i += gridDim.x*blockDim.x){
    float v = p[i]; s += v; q += v*v;
  }
  ls[threadIdx.x] = s; lq[threadIdx.x] = q;
  __syncthreads();
  for (int o = 128; o > 0; o >>= 1){
    if ((int)threadIdx.x < o){ ls[threadIdx.x] += ls[threadIdx.x+o]; lq[threadIdx.x] += lq[threadIdx.x+o]; }
    __syncthreads();
  }
  if (threadIdx.x == 0){
    atomicAdd(&sums[b*2+0], ls[0]);
    atomicAdd(&sums[b*2+1], lq[0]);
  }
}

__global__ void k_stats_final(const float* __restrict__ sums, float* __restrict__ st, float invN){
  int b = threadIdx.x;
  if (b < BB){
    float m = sums[2*b]*invN;
    float v = sums[2*b+1]*invN - m*m;
    st[2*b]   = m;
    st[2*b+1] = rsqrtf(v + 1e-5f);
  }
}

// ---------------------------------------------------------------- lift: 6 -> gelu(128) -> 32 (+inj copy)
__global__ void k_lift(const float* __restrict__ x,  const float* __restrict__ w1,
                       const float* __restrict__ b1, const float* __restrict__ w2,
                       const float* __restrict__ b2, float* __restrict__ xo,
                       float* __restrict__ inj){
  __shared__ bf16 hb[8][16][HID];
  const int lane = threadIdx.x & 31, wv = threadIdx.x >> 5;
  const int h = lane >> 4, nc = lane & 15;
  const int tile = blockIdx.x*8 + wv;               // 16-voxel tile
  const int b  = tile >> 14;                        // N3/16 = 16384 tiles per batch
  const int p0 = (tile & 16383) * 16;

  v16bf a;
  #pragma unroll
  for (int e = 0; e < 16; ++e){
    int k = akk(lane, e);
    a[e] = (k < CIN) ? (bf16)x[(b*CIN + k)*N3 + p0 + nc] : (bf16)0.0f;
  }
  #pragma unroll
  for (int nt = 0; nt < 8; ++nt){
    v16bf w;
    #pragma unroll
    for (int e = 0; e < 16; ++e){
      int k = (h << 4) + e;
      w[e] = (k < CIN) ? (bf16)w1[(nt*16 + nc)*CIN + k] : (bf16)0.0f;
    }
    v8f c = {};
    c = WMMA(a, w, c);
    float bias = b1[nt*16 + nc];
    #pragma unroll
    for (int e = 0; e < 8; ++e) hb[wv][h*8 + e][nt*16 + nc] = (bf16)gelu_f(c[e] + bias);
  }
  __syncthreads();
  #pragma unroll
  for (int nt = 0; nt < 2; ++nt){
    v8f acc = {};
    #pragma unroll
    for (int ks = 0; ks < 4; ++ks){
      v16bf a2, w;
      #pragma unroll
      for (int e = 0; e < 16; ++e) a2[e] = hb[wv][nc][ks*32 + akk(lane, e)];
      #pragma unroll
      for (int e = 0; e < 16; ++e) w[e] = (bf16)w2[(nt*16 + nc)*HID + ks*32 + (h << 4) + e];
      acc = WMMA(a2, w, acc);
    }
    float bias = b2[nt*16 + nc];
    #pragma unroll
    for (int e = 0; e < 8; ++e){
      int o  = nt*16 + nc;
      int oi = (b*CM + o)*N3 + p0 + h*8 + e;
      float v = acc[e] + bias;
      xo[oi] = v; inj[oi] = v;
    }
  }
}

// ---------------------------------------------------------------- forward truncated DFT, z axis
// in : x [b,c,xx,yy,z] (normalized on load, async-staged through LDS)
// out: A1[row=(b,c,xx,yy)][kz<16] complex
__global__ void k_fwdz(const float* __restrict__ x, const float* __restrict__ st,
                       float2* __restrict__ A1){
  // single shared object -> LDS offset 0; xs starts at byte 256
  __shared__ struct { float ct[64]; float xs[8][16][64]; } sm;
  if (threadIdx.x < 64) sm.ct[threadIdx.x] = cosf(TWO_PI_64*(float)threadIdx.x);
  __syncthreads();
  const int lane = threadIdx.x & 31, wv = threadIdx.x >> 5;
  const int h = lane >> 4, nc = lane & 15;

  // tile-invariant twiddles: B[k=z][n=kz], e^{-i th}
  v16bf tc[2], tsn[2];
  #pragma unroll
  for (int ks = 0; ks < 2; ++ks)
    #pragma unroll
    for (int e = 0; e < 16; ++e){
      int z = ks*32 + (h << 4) + e;
      int p = (nc*z) & 63;
      tc[ks][e]  = (bf16)sm.ct[p];              // cos
      tsn[ks][e] = (bf16)sm.ct[(p+16)&63];      // -sin
    }

  const unsigned ldsbase = 256u + (unsigned)wv*4096u;   // bytes into sm.xs for this wave
  for (int t = 0; t < 8; ++t){
    const int tile = (blockIdx.x*8 + wv)*8 + t;         // 131072 tiles
    const int r0 = tile*16;
    const int b  = r0 >> 17;
    const float mean = st[2*b], rs = st[2*b+1];

    // async-stage 16 rows x 64 floats (4 KB) into this wave's LDS slice
    #pragma unroll
    for (int i = 0; i < 8; ++i){
      int ch = i*2 + h;                                 // 16B chunk id, 0..15
      unsigned loff = ldsbase + (unsigned)nc*256u + (unsigned)ch*16u;
      unsigned long long ga = (unsigned long long)(const void*)(x + (size_t)(r0 + nc)*64 + ch*4);
      asm volatile("global_load_async_to_lds_b128 %0, %1, off"
                   :: "v"(loff), "v"(ga) : "memory");
    }
    asm volatile("s_wait_asynccnt 0x0" ::: "memory");

    v8f dre = {}, di = {};
    #pragma unroll
    for (int ks = 0; ks < 2; ++ks){
      v16bf a;
      #pragma unroll
      for (int e = 0; e < 16; ++e){
        int z = ks*32 + akk(lane, e);
        a[e] = (bf16)((sm.xs[wv][nc][z] - mean)*rs);
      }
      dre = WMMA(a, tc[ks],  dre);
      di  = WMMA(a, tsn[ks], di);
    }
    #pragma unroll
    for (int e = 0; e < 8; ++e)
      A1[(r0 + h*8 + e)*16 + nc] = make_float2(dre[e], di[e]);
  }
}

// ---------------------------------------------------------------- forward DFT, y axis (ky in {0..15,48..63})
// in: A1[(bc,xx)][y][kz]  out: A2[bc][j][kz][xx]
__global__ void k_fwdy(const float2* __restrict__ A1, float2* __restrict__ A2){
  __shared__ float ct[64];
  if (threadIdx.x < 64) ct[threadIdx.x] = cosf(TWO_PI_64*(float)threadIdx.x);
  __syncthreads();
  const int lane = threadIdx.x & 31, wv = threadIdx.x >> 5;
  const int h = lane >> 4, nc = lane & 15;

  v16bf tc[2][2], ts[2][2];                     // [nt][ks]
  #pragma unroll
  for (int nt = 0; nt < 2; ++nt)
    #pragma unroll
    for (int ks = 0; ks < 2; ++ks)
      #pragma unroll
      for (int e = 0; e < 16; ++e){
        int y  = ks*32 + (h << 4) + e;
        int j  = nt*16 + nc;
        int ky = (j < 16) ? j : j + 32;
        int p  = (ky*y) & 63;
        tc[nt][ks][e] = (bf16)ct[p];
        ts[nt][ks][e] = (bf16)ct[(p+48)&63];    // sin
      }

  for (int t = 0; t < 4; ++t){
    const int bcx = (blockIdx.x*8 + wv)*4 + t;  // bc*64 + xx, 16384 tiles
    const int base = bcx*1024;
    v16bf ar[2], ai[2], nar[2];
    #pragma unroll
    for (int ks = 0; ks < 2; ++ks)
      #pragma unroll
      for (int e = 0; e < 16; ++e){
        int y = ks*32 + akk(lane, e);
        float2 v = A1[base + y*16 + nc];        // row = kz = nc
        ar[ks][e] = (bf16)v.x; ai[ks][e] = (bf16)v.y; nar[ks][e] = (bf16)(-v.x);
      }
    #pragma unroll
    for (int nt = 0; nt < 2; ++nt){
      v8f dre = {}, di = {};
      #pragma unroll
      for (int ks = 0; ks < 2; ++ks){
        dre = WMMA(ar[ks],  tc[nt][ks], dre);   // e^{-i th}: re = ar*c + ai*s
        dre = WMMA(ai[ks],  ts[nt][ks], dre);
        di  = WMMA(ai[ks],  tc[nt][ks], di);    // im = ai*c - ar*s
        di  = WMMA(nar[ks], ts[nt][ks], di);
      }
      #pragma unroll
      for (int e = 0; e < 8; ++e){
        int kz = h*8 + e;
        int j  = nt*16 + nc;
        A2[(((bcx >> 6)*32 + j)*16 + kz)*64 + (bcx & 63)] = make_float2(dre[e], di[e]);
      }
    }
  }
}

// ---------------------------------------------------------------- forward DFT, x axis -> corner spectrum
// in: A2[bc][j][kz][xx]  out: Xf[bc][mode], mode = i*512 + j*16 + kz
__global__ void k_fwdx(const float2* __restrict__ A2, float2* __restrict__ Xf){
  __shared__ float ct[64];
  if (threadIdx.x < 64) ct[threadIdx.x] = cosf(TWO_PI_64*(float)threadIdx.x);
  __syncthreads();
  const int lane = threadIdx.x & 31, wv = threadIdx.x >> 5;
  const int h = lane >> 4, nc = lane & 15;

  v16bf tc[2][2], ts[2][2];
  #pragma unroll
  for (int nt = 0; nt < 2; ++nt)
    #pragma unroll
    for (int ks = 0; ks < 2; ++ks)
      #pragma unroll
      for (int e = 0; e < 16; ++e){
        int xx = ks*32 + (h << 4) + e;
        int i  = nt*16 + nc;
        int kx = (i < 16) ? i : i + 32;
        int p  = (kx*xx) & 63;
        tc[nt][ks][e] = (bf16)ct[p];
        ts[nt][ks][e] = (bf16)ct[(p+48)&63];
      }

  for (int t = 0; t < 4; ++t){
    const int tile = (blockIdx.x*8 + wv)*4 + t;  // 8192 tiles, fixed (bc, j)
    const int bc = tile >> 5, j = tile & 31;
    const int base = ((bc*32 + j)*16)*64;
    v16bf ar[2], ai[2], nar[2];
    #pragma unroll
    for (int ks = 0; ks < 2; ++ks)
      #pragma unroll
      for (int e = 0; e < 16; ++e){
        int xx = ks*32 + akk(lane, e);
        float2 v = A2[base + nc*64 + xx];        // row = kz = nc
        ar[ks][e] = (bf16)v.x; ai[ks][e] = (bf16)v.y; nar[ks][e] = (bf16)(-v.x);
      }
    #pragma unroll
    for (int nt = 0; nt < 2; ++nt){
      v8f dre = {}, di = {};
      #pragma unroll
      for (int ks = 0; ks < 2; ++ks){
        dre = WMMA(ar[ks],  tc[nt][ks], dre);
        dre = WMMA(ai[ks],  ts[nt][ks], dre);
        di  = WMMA(ai[ks],  tc[nt][ks], di);
        di  = WMMA(nar[ks], ts[nt][ks], di);
      }
      #pragma unroll
      for (int e = 0; e < 8; ++e){
        int kz = h*8 + e;
        int i  = nt*16 + nc;
        Xf[bc*16384 + i*512 + j*16 + kz] = make_float2(dre[e], di[e]);
      }
    }
  }
}

// ---------------------------------------------------------------- spectral channel mixing (per mode, complex)
__global__ void k_spec(const float2* __restrict__ Xf, const float* __restrict__ wre,
                       const float* __restrict__ wim, float2* __restrict__ Yf, int l){
  const int lane = threadIdx.x & 31, wv = threadIdx.x >> 5;
  const int h = lane >> 4, nc = lane & 15;
  const int mode = blockIdx.x*8 + wv;               // 16384 modes
  const int i_kx = mode >> 9, j_ky = (mode >> 4) & 31, kz = mode & 15;
  const int q  = ((i_kx >= 16) ? 1 : 0) + ((j_ky >= 16) ? 2 : 0);
  const int mx = i_kx & 15, my = j_ky & 15;
  const int wbase = (l*4 + q)*CM*CM;                // + i*CM + o, then *4096 + mof
  const int mof   = mx*256 + my*16 + kz;
  __builtin_prefetch(&wre[wbase*4096 + mof], 0, 1);
  __builtin_prefetch(&wim[wbase*4096 + mof], 0, 1);

  v16bf ar, ai, nai;                                // A: rows = batch (8 valid of 16), K = ci
  #pragma unroll
  for (int e = 0; e < 16; ++e){
    int ci = akk(lane, e);
    if (nc < 8){
      float2 v = Xf[(nc*CM + ci)*16384 + mode];
      ar[e] = (bf16)v.x; ai[e] = (bf16)v.y; nai[e] = (bf16)(-v.y);
    } else { ar[e] = (bf16)0.0f; ai[e] = (bf16)0.0f; nai[e] = (bf16)0.0f; }
  }
  #pragma unroll
  for (int nt = 0; nt < 2; ++nt){
    v16bf br, bi;
    #pragma unroll
    for (int e = 0; e < 16; ++e){
      int ci = (h << 4) + e;
      int o  = nt*16 + nc;
      int wi = (wbase + ci*CM + o)*4096 + mof;
      br[e] = (bf16)wre[wi]; bi[e] = (bf16)wim[wi];
    }
    v8f dre = {}, di = {};
    dre = WMMA(ar,  br, dre);                       // re = Xr*Wr - Xi*Wi
    dre = WMMA(nai, bi, dre);
    di  = WMMA(ar,  bi, di);                        // im = Xr*Wi + Xi*Wr
    di  = WMMA(ai,  br, di);
    #pragma unroll
    for (int e = 0; e < 8; ++e){
      int b = h*8 + e;
      if (b < 8){
        int o = nt*16 + nc;
        Yf[(b*CM + o)*16384 + mode] = make_float2(dre[e], di[e]);
      }
    }
  }
}

// ---------------------------------------------------------------- inverse DFT, x axis
// in: Yf[bo][mode]  out: B2[bo][xx][kz][j]
__global__ void k_invx(const float2* __restrict__ Yf, float2* __restrict__ B2){
  __shared__ float ct[64];
  if (threadIdx.x < 64) ct[threadIdx.x] = cosf(TWO_PI_64*(float)threadIdx.x);
  __syncthreads();
  const int lane = threadIdx.x & 31, wv = threadIdx.x >> 5;
  const int h = lane >> 4, nc = lane & 15;

  v16bf tc[4], ts[4];                               // [nt], K = i_kx
  #pragma unroll
  for (int nt = 0; nt < 4; ++nt)
    #pragma unroll
    for (int e = 0; e < 16; ++e){
      int ik = (h << 4) + e;
      int kx = (ik < 16) ? ik : ik - 32;
      int xx = nt*16 + nc;
      int p  = (kx*xx) & 63;
      tc[nt][e] = (bf16)ct[p];
      ts[nt][e] = (bf16)ct[(p+48)&63];
    }

  for (int t = 0; t < 4; ++t){
    const int tile = (blockIdx.x*8 + wv)*4 + t;     // 8192 tiles, fixed (bo, j)
    const int bo = tile >> 5, j = tile & 31;
    v16bf yr, yi, nyi;                              // rows = kz
    #pragma unroll
    for (int e = 0; e < 16; ++e){
      int ik = akk(lane, e);
      float2 v = Yf[bo*16384 + ik*512 + j*16 + nc];
      yr[e] = (bf16)v.x; yi[e] = (bf16)v.y; nyi[e] = (bf16)(-v.y);
    }
    #pragma unroll
    for (int nt = 0; nt < 4; ++nt){
      v8f dre = {}, di = {};
      dre = WMMA(yr,  tc[nt], dre);                 // e^{+i th}: re = Yr*c - Yi*s
      dre = WMMA(nyi, ts[nt], dre);
      di  = WMMA(yr,  ts[nt], di);                  // im = Yr*s + Yi*c
      di  = WMMA(yi,  tc[nt], di);
      #pragma unroll
      for (int e = 0; e < 8; ++e){
        int kz = h*8 + e, xx = nt*16 + nc;
        B2[((bo*64 + xx)*16 + kz)*32 + j] = make_float2(dre[e], di[e]);
      }
    }
  }
}

// ---------------------------------------------------------------- inverse DFT, y axis
// in: B2[(bo,xx)][kz][j]  out: B1[(bo,xx)][yy][kz]
__global__ void k_invy(const float2* __restrict__ B2, float2* __restrict__ B1){
  __shared__ float ct[64];
  if (threadIdx.x < 64) ct[threadIdx.x] = cosf(TWO_PI_64*(float)threadIdx.x);
  __syncthreads();
  const int lane = threadIdx.x & 31, wv = threadIdx.x >> 5;
  const int h = lane >> 4, nc = lane & 15;

  v16bf tc[4], ts[4];                               // [nt], K = j
  #pragma unroll
  for (int nt = 0; nt < 4; ++nt)
    #pragma unroll
    for (int e = 0; e < 16; ++e){
      int jk = (h << 4) + e;
      int ky = (jk < 16) ? jk : jk - 32;
      int yy = nt*16 + nc;
      int p  = (ky*yy) & 63;
      tc[nt][e] = (bf16)ct[p];
      ts[nt][e] = (bf16)ct[(p+48)&63];
    }

  for (int t = 0; t < 4; ++t){
    const int box = (blockIdx.x*8 + wv)*4 + t;      // bo*64 + xx, 16384 tiles
    const int base = box*512;
    v16bf yr, yi, nyi;                              // rows = kz
    #pragma unroll
    for (int e = 0; e < 16; ++e){
      int jk = akk(lane, e);
      float2 v = B2[base + nc*32 + jk];
      yr[e] = (bf16)v.x; yi[e] = (bf16)v.y; nyi[e] = (bf16)(-v.y);
    }
    #pragma unroll
    for (int nt = 0; nt < 4; ++nt){
      v8f dre = {}, di = {};
      dre = WMMA(yr,  tc[nt], dre);
      dre = WMMA(nyi, ts[nt], dre);
      di  = WMMA(yr,  ts[nt], di);
      di  = WMMA(yi,  tc[nt], di);
      #pragma unroll
      for (int e = 0; e < 8; ++e){
        int kz = h*8 + e, yy = nt*16 + nc;
        B1[(box*64 + yy)*16 + kz] = make_float2(dre[e], di[e]);
      }
    }
  }
}

// ---------------------------------------------------------------- inverse rfft, z axis (real out, Hermitian fold)
// in: B1[row=(bo,xx,yy)][kz]  out: x1[row][z]
__global__ void k_invz(const float2* __restrict__ B1, float* __restrict__ x1){
  __shared__ float ct[64];
  if (threadIdx.x < 64) ct[threadIdx.x] = cosf(TWO_PI_64*(float)threadIdx.x);
  __syncthreads();
  const int lane = threadIdx.x & 31, wv = threadIdx.x >> 5;
  const int h = lane >> 4, nc = lane & 15;

  v16bf tcs[4], tsn[4];                             // [nt], K = kz (16 valid of 32), scaled
  #pragma unroll
  for (int nt = 0; nt < 4; ++nt)
    #pragma unroll
    for (int e = 0; e < 16; ++e){
      int k = (h << 4) + e;
      int z = nt*16 + nc;
      if (k < 16){
        float sc = ((k == 0) ? 1.0f : 2.0f) * INV_VOL;
        int p = (k*z) & 63;
        tcs[nt][e] = (bf16)(sc*ct[p]);              // sc*cos
        tsn[nt][e] = (bf16)(sc*ct[(p+16)&63]);      // -sc*sin
      } else { tcs[nt][e] = (bf16)0.0f; tsn[nt][e] = (bf16)0.0f; }
    }

  for (int t = 0; t < 8; ++t){
    const int tile = (blockIdx.x*8 + wv)*8 + t;     // 65536 tiles
    const int r0 = tile*16;
    v16bf ar, ai;
    #pragma unroll
    for (int e = 0; e < 16; ++e){
      int k = akk(lane, e);
      if (k < 16){
        float2 v = B1[(r0 + nc)*16 + k];
        ar[e] = (bf16)v.x; ai[e] = (bf16)v.y;
      } else { ar[e] = (bf16)0.0f; ai[e] = (bf16)0.0f; }
    }
    #pragma unroll
    for (int nt = 0; nt < 4; ++nt){
      v8f d = {};
      d = WMMA(ar, tcs[nt], d);                     // x = sum s_k (Xr cos - Xi sin)/64^3
      d = WMMA(ai, tsn[nt], d);
      #pragma unroll
      for (int e = 0; e < 8; ++e)
        x1[(r0 + h*8 + e)*64 + nt*16 + nc] = d[e];
    }
  }
}

// ---------------------------------------------------------------- 1x1x1 conv on normalized x
__global__ void k_filt(const float* __restrict__ x, const float* __restrict__ st,
                       const float* __restrict__ fw, const float* __restrict__ fb,
                       float* __restrict__ x2, int l){
  const int lane = threadIdx.x & 31, wv = threadIdx.x >> 5;
  const int h = lane >> 4, nc = lane & 15;
  const int tile = blockIdx.x*8 + wv;
  const int b  = tile >> 14;
  const int p0 = (tile & 16383)*16;
  const float mean = st[2*b], rs = st[2*b+1];

  v16bf a;
  #pragma unroll
  for (int e = 0; e < 16; ++e){
    int c = akk(lane, e);
    a[e] = (bf16)((x[(b*CM + c)*N3 + p0 + nc] - mean)*rs);
  }
  #pragma unroll
  for (int nt = 0; nt < 2; ++nt){
    v16bf w;
    #pragma unroll
    for (int e = 0; e < 16; ++e)
      w[e] = (bf16)fw[l*CM*CM + (nt*16 + nc)*CM + (h << 4) + e];
    v8f c = {};
    c = WMMA(a, w, c);
    float bias = fb[l*CM + nt*16 + nc];
    #pragma unroll
    for (int e = 0; e < 8; ++e)
      x2[(b*CM + nt*16 + nc)*N3 + p0 + h*8 + e] = c[e] + bias;
  }
}

// ---------------------------------------------------------------- residual combine: x = gelu(gn(x1)+gn(x2)+inj) + x
__global__ void k_combine(float* __restrict__ x,  const float* __restrict__ x1,
                          const float* __restrict__ x2, const float* __restrict__ inj,
                          const float* __restrict__ s1, const float* __restrict__ s2){
  const long long total = (long long)BB*PERB;
  for (long long i = (long long)blockIdx.x*blockDim.x + threadIdx.x; i < total;
       i += (long long)gridDim.x*blockDim.x){
    int b = (int)(i >> 23);                         // / 8388608
    float v = (x1[i] - s1[2*b])*s1[2*b+1] + (x2[i] - s2[2*b])*s2[2*b+1] + inj[i];
    x[i] = gelu_f(v) + x[i];
  }
}

// ---------------------------------------------------------------- proj: 32 -> gelu(128) -> 6 (no final bias)
__global__ void k_proj(const float* __restrict__ x,  const float* __restrict__ w1,
                       const float* __restrict__ b1, const float* __restrict__ w2,
                       float* __restrict__ out){
  __shared__ bf16 hb[8][16][HID];
  const int lane = threadIdx.x & 31, wv = threadIdx.x >> 5;
  const int h = lane >> 4, nc = lane & 15;
  const int tile = blockIdx.x*8 + wv;
  const int b  = tile >> 14;
  const int p0 = (tile & 16383)*16;

  v16bf a;
  #pragma unroll
  for (int e = 0; e < 16; ++e)
    a[e] = (bf16)x[(b*CM + akk(lane, e))*N3 + p0 + nc];
  #pragma unroll
  for (int nt = 0; nt < 8; ++nt){
    v16bf w;
    #pragma unroll
    for (int e = 0; e < 16; ++e)
      w[e] = (bf16)w1[(nt*16 + nc)*CM + (h << 4) + e];
    v8f c = {};
    c = WMMA(a, w, c);
    float bias = b1[nt*16 + nc];
    #pragma unroll
    for (int e = 0; e < 8; ++e) hb[wv][h*8 + e][nt*16 + nc] = (bf16)gelu_f(c[e] + bias);
  }
  __syncthreads();
  v8f acc = {};
  #pragma unroll
  for (int ks = 0; ks < 4; ++ks){
    v16bf a2, w;
    #pragma unroll
    for (int e = 0; e < 16; ++e) a2[e] = hb[wv][nc][ks*32 + akk(lane, e)];
    #pragma unroll
    for (int e = 0; e < 16; ++e)
      w[e] = (nc < COUT) ? (bf16)w2[nc*HID + ks*32 + (h << 4) + e] : (bf16)0.0f;
    acc = WMMA(a2, w, acc);
  }
  if (nc < COUT){
    #pragma unroll
    for (int e = 0; e < 8; ++e)
      out[(b*COUT + nc)*N3 + p0 + h*8 + e] = acc[e];
  }
}

// ---------------------------------------------------------------- host launcher
extern "C" void kernel_launch(void* const* d_in, const int* in_sizes, int n_in,
                              void* d_out, int out_size, void* d_ws, size_t ws_size,
                              hipStream_t stream){
  (void)in_sizes; (void)n_in; (void)out_size; (void)ws_size;
  const float* x    = (const float*)d_in[0];
  const float* lw1  = (const float*)d_in[1];
  const float* lb1  = (const float*)d_in[2];
  const float* lw2  = (const float*)d_in[3];
  const float* lb2  = (const float*)d_in[4];
  const float* swre = (const float*)d_in[5];
  const float* swim = (const float*)d_in[6];
  const float* fw   = (const float*)d_in[7];
  const float* fb   = (const float*)d_in[8];
  const float* pw1  = (const float*)d_in[9];
  const float* pb1  = (const float*)d_in[10];
  const float* pw2  = (const float*)d_in[11];

  float* ws = (float*)d_ws;
  const size_t EV = (size_t)BB*PERB;                // 67,108,864 floats
  float*  Xc  = ws;
  float*  Inj = Xc  + EV;
  float*  X1  = Inj + EV;
  float*  X2  = X1  + EV;
  float2* A1  = (float2*)(X2 + EV);                 // 16,777,216 complex
  float2* A2  = (float2*)((float*)A1 + 33554432ull);// 8,388,608 complex
  float2* Xf  = (float2*)((float*)A2 + 16777216ull);// 4,194,304 complex
  float2* Yf  = (float2*)((float*)Xf +  8388608ull);// 4,194,304 complex
  float*  sums = (float*)Yf + 8388608ull;           // 48 floats
  float*  st   = sums + 64;                         // 48 floats

  const dim3 t256(256);
  const float invN = 1.0f/(float)PERB;

  k_lift<<<16384, t256, 0, stream>>>(x, lw1, lb1, lw2, lb2, Xc, Inj);

  for (int l = 0; l < NL; ++l){
    k_zero<<<1, 64, 0, stream>>>(sums, 48);
    k_stats_partial<<<dim3(512, BB), t256, 0, stream>>>(Xc, sums + 0, PERB);
    k_stats_final<<<1, BB, 0, stream>>>(sums + 0, st + 0, invN);

    k_fwdz<<<2048, t256, 0, stream>>>(Xc, st + 0, A1);
    k_fwdy<<<512,  t256, 0, stream>>>(A1, A2);
    k_fwdx<<<256,  t256, 0, stream>>>(A2, Xf);
    k_spec<<<2048, t256, 0, stream>>>(Xf, swre, swim, Yf, l);
    k_invx<<<256,  t256, 0, stream>>>(Yf, A2);
    k_invy<<<512,  t256, 0, stream>>>(A2, A1);
    k_invz<<<1024, t256, 0, stream>>>(A1, X1);

    k_stats_partial<<<dim3(512, BB), t256, 0, stream>>>(X1, sums + 16, PERB);
    k_stats_final<<<1, BB, 0, stream>>>(sums + 16, st + 16, invN);

    k_filt<<<16384, t256, 0, stream>>>(Xc, st + 0, fw, fb, X2, l);
    k_stats_partial<<<dim3(512, BB), t256, 0, stream>>>(X2, sums + 32, PERB);
    k_stats_final<<<1, BB, 0, stream>>>(sums + 32, st + 32, invN);

    k_combine<<<16384, t256, 0, stream>>>(Xc, X1, X2, Inj, st + 16, st + 32);
  }

  k_proj<<<16384, t256, 0, stream>>>(Xc, pw1, pb1, pw2, (float*)d_out);
}
